// ContextGuidedAdaptiveAttention_81406810129141
// MI455X (gfx1250) — matrix operand order repaired and checked by reading, not compiled, not run on hardware
//
#include <hip/hip_runtime.h>
#include <hip/hip_bf16.h>

typedef __attribute__((ext_vector_type(16))) _Float16 v16h;
typedef __attribute__((ext_vector_type(8)))  _Float16 v8h;
typedef __attribute__((ext_vector_type(8)))  float    v8f;

#define BB   16
#define CC   256
#define HH   56
#define WW2  56
#define NN   3136            // 56*56
#define NHD  8               // heads
#define PP   4               // sample points
#define HD   32              // head dim
#define SCALE_F 0.17677669529663687f  // 32^-0.5

__device__ __forceinline__ float gelu_f(float x) {
    return 0.5f * x * (1.0f + erff(x * 0.70710678118654752f));
}

// ---------------------------------------------------------------------------
// WMMA GEMM:  Y[b,o,n] = epilogue( sum_c  Wm[o,c] * X[b,c,n] )
// M = 256 (o), K = 256 (c), N = 3136 (n), 16 batches.
// Block: 256 threads (8 waves, 4x2), 64x64 output tile, K staged in 32-chunks.
// Software pipelined: next tile's global loads issued before current tile's
// WMMAs; LDS tiles double-buffered -> one barrier per K-step.
// EPI: 0=none, 1=GELU, 2=BatchNorm(p0=mean,p1=gamma,p2=var,p3=beta)
// ---------------------------------------------------------------------------
template<int EPI, typename OUT_T>
__global__ __launch_bounds__(256) void gemm256(
    const float* __restrict__ X,
    const float* __restrict__ Wm,
    OUT_T* __restrict__ Y,
    const float* __restrict__ p0, const float* __restrict__ p1,
    const float* __restrict__ p2, const float* __restrict__ p3)
{
    constexpr int GK = 256, GM = 256, GN = NN;
    constexpr int TILE = 64 * 40;         // halfs per LDS buffer
    const int bM = blockIdx.x;            // 0..3
    const int bN = blockIdx.y;            // 0..48
    const int b  = blockIdx.z;            // 0..15
    const float* Xb = X + (size_t)b * GK * GN;

    __shared__ _Float16 lA[2 * TILE];     // A tile [m 0..63][k 0..31], stride 40 halfs (80B)
    __shared__ _Float16 lB[2 * TILE];     // B tile transposed [n 0..63][k 0..31]

    const int t    = threadIdx.x;
    const int lane = t & 31;
    const int wave = t >> 5;
    const int wm   = wave >> 1;           // 0..3 : 16-row group
    const int wn   = wave & 1;            // 0..1 : 32-col group
    const int lm   = lane & 15;
    const int lh   = lane >> 4;

    // per-thread staging coordinates
    const int ar = t >> 3, ac = t & 7;    // A rows ar, ar+32 ; cols ac*4..+3
    const int br = t >> 4, bc = t & 15;   // B rows br, br+16 ; cols bc*4..+3

    v8f acc0 = {0.f,0.f,0.f,0.f,0.f,0.f,0.f,0.f};
    v8f acc1 = {0.f,0.f,0.f,0.f,0.f,0.f,0.f,0.f};

    float4 rA0, rA1, rB0, rB1;

    auto load_tile = [&](int k0) {
        rA0 = *(const float4*)(Wm + (size_t)(bM*64 + ar)      * GK + k0 + ac*4);
        rA1 = *(const float4*)(Wm + (size_t)(bM*64 + ar + 32) * GK + k0 + ac*4);
        rB0 = *(const float4*)(Xb + (size_t)(k0 + br)      * GN + bN*64 + bc*4);
        rB1 = *(const float4*)(Xb + (size_t)(k0 + br + 16) * GN + bN*64 + bc*4);
    };
    auto store_tile = [&](int buf) {
        _Float16* A = lA + buf * TILE;
        _Float16* d0 = &A[ar*40 + ac*4];
        d0[0] = (_Float16)rA0.x; d0[1] = (_Float16)rA0.y;
        d0[2] = (_Float16)rA0.z; d0[3] = (_Float16)rA0.w;
        _Float16* d1 = &A[(ar+32)*40 + ac*4];
        d1[0] = (_Float16)rA1.x; d1[1] = (_Float16)rA1.y;
        d1[2] = (_Float16)rA1.z; d1[3] = (_Float16)rA1.w;
        _Float16* Bt = lB + buf * TILE;
        Bt[(bc*4+0)*40 + br]      = (_Float16)rB0.x;
        Bt[(bc*4+1)*40 + br]      = (_Float16)rB0.y;
        Bt[(bc*4+2)*40 + br]      = (_Float16)rB0.z;
        Bt[(bc*4+3)*40 + br]      = (_Float16)rB0.w;
        Bt[(bc*4+0)*40 + br + 16] = (_Float16)rB1.x;
        Bt[(bc*4+1)*40 + br + 16] = (_Float16)rB1.y;
        Bt[(bc*4+2)*40 + br + 16] = (_Float16)rB1.z;
        Bt[(bc*4+3)*40 + br + 16] = (_Float16)rB1.w;
    };

    // prologue: stage K-tile 0 into buffer 0
    load_tile(0);
    store_tile(0);

    int cur = 0;
#pragma unroll
    for (int k0 = 0; k0 < GK; k0 += 32) {
        __syncthreads();                          // buf[cur] ready for all waves
        const bool has_next = (k0 + 32 < GK);
        if (has_next) {
            load_tile(k0 + 32);                   // 4 global_load_b128 in flight,
                                                  // hidden under fragment loads + WMMA
            if (k0 + 64 < GK) {                   // prefetch tile after next (L2)
                __builtin_prefetch(Wm + (size_t)(bM*64 + ar)*GK + (k0+64) + ac*4, 0, 3);
                __builtin_prefetch(Xb + (size_t)(k0+64 + br)*GN + bN*64 + bc*4, 0, 3);
            }
        }

        // ---- A fragment: lane holds row M=lm; e<8 -> k=lh*8+e, e>=8 -> k=16+lh*8+(e-8)
        const _Float16* arow = &lA[cur*TILE + (wm*16 + lm)*40 + lh*8];
        v8h alo = *(const v8h*)(arow);
        v8h ahi = *(const v8h*)(arow + 16);
        v16h af;
#pragma unroll
        for (int i = 0; i < 8; ++i) { af[i] = alo[i]; af[8+i] = ahi[i]; }

        // ---- B fragments: lane holds col N=lm; e<8 -> k=lh*8+e, e>=8 -> k=16+lh*8+(e-8)
        {
            const _Float16* brow = &lB[cur*TILE + (wn*32 + lm)*40 + lh*8];
            v8h b0 = *(const v8h*)(brow);
            v8h b1 = *(const v8h*)(brow + 16);
            v16h bf;
#pragma unroll
            for (int i = 0; i < 8; ++i) { bf[i] = b0[i]; bf[8+i] = b1[i]; }
            acc0 = __builtin_amdgcn_wmma_f32_16x16x32_f16(false, af, false, bf,
                                                          (short)0, acc0, false, false);
        }
        {
            const _Float16* brow = &lB[cur*TILE + (wn*32 + 16 + lm)*40 + lh*8];
            v8h b0 = *(const v8h*)(brow);
            v8h b1 = *(const v8h*)(brow + 16);
            v16h bf;
#pragma unroll
            for (int i = 0; i < 8; ++i) { bf[i] = b0[i]; bf[8+i] = b1[i]; }
            acc1 = __builtin_amdgcn_wmma_f32_16x16x32_f16(false, af, false, bf,
                                                          (short)0, acc1, false, false);
        }

        if (has_next) store_tile(1 - cur);        // writes other buffer: no barrier vs
        cur ^= 1;                                 // concurrent reads of buf[cur]
    }

    // ---- epilogue + store. D layout: VGPR v -> M = v + 8*lh, N = lm
    const int ncol = bN*64 + wn*32 + lm;
#pragma unroll
    for (int v = 0; v < 8; ++v) {
        const int o = bM*64 + wm*16 + lh*8 + v;
        float x0 = acc0[v];
        float x1 = acc1[v];
        if (EPI == 1) {
            x0 = gelu_f(x0); x1 = gelu_f(x1);
        } else if (EPI == 2) {
            const float inv = p1[o] * rsqrtf(p2[o] + 1e-5f);
            x0 = (x0 - p0[o]) * inv + p3[o];
            x1 = (x1 - p0[o]) * inv + p3[o];
        }
        const size_t rb = ((size_t)b*GM + o)*GN;
        Y[rb + ncol]      = (OUT_T)x0;
        Y[rb + ncol + 16] = (OUT_T)x1;
    }
}

// ---------------------------------------------------------------------------
// 8x8 average pool: [B,256,56,56] -> [B,256,7,7]
// ---------------------------------------------------------------------------
__global__ __launch_bounds__(256) void pool8x8(const float* __restrict__ x,
                                               float* __restrict__ y)
{
    const int tid = blockIdx.x*256 + threadIdx.x;
    const int pos = tid % 49;
    const int c   = (tid / 49) % CC;
    const int b   = tid / (49*CC);
    const int ph = pos / 7, pw = pos % 7;
    const float* xb = x + ((size_t)b*CC + c)*NN;
    float s = 0.f;
#pragma unroll
    for (int iy = 0; iy < 8; ++iy)
#pragma unroll
        for (int ix = 0; ix < 8; ++ix)
            s += xb[(ph*8+iy)*WW2 + pw*8+ix];
    y[((size_t)b*CC + c)*49 + pos] = s * (1.0f/64.0f);
}

// ---------------------------------------------------------------------------
// Channel LN over 256 + 32x256 conv on the 7x7 map. One block per (b,pos).
// ---------------------------------------------------------------------------
__global__ __launch_bounds__(256) void cg_ln_conv(const float* __restrict__ pooled,
                                                  const float* __restrict__ g,
                                                  const float* __restrict__ be,
                                                  const float* __restrict__ w,
                                                  float* __restrict__ t3)
{
    const int pos = blockIdx.x % 49;
    const int b   = blockIdx.x / 49;
    const int c   = threadIdx.x;
    __shared__ float red[256];
    __shared__ float av[256];
    const float x = pooled[((size_t)b*CC + c)*49 + pos];
    red[c] = x; __syncthreads();
    for (int s = 128; s > 0; s >>= 1) { if (c < s) red[c] += red[c+s]; __syncthreads(); }
    const float mu = red[0] * (1.0f/256.0f);
    __syncthreads();
    const float d = x - mu;
    red[c] = d*d; __syncthreads();
    for (int s = 128; s > 0; s >>= 1) { if (c < s) red[c] += red[c+s]; __syncthreads(); }
    const float var = red[0] * (1.0f/256.0f);
    av[c] = d * rsqrtf(var + 1e-6f) * g[c] + be[c];
    __syncthreads();
    if (c < 32) {
        float o = 0.f;
        for (int i = 0; i < 256; ++i) o += w[c*256 + i] * av[i];
        t3[((size_t)b*32 + c)*49 + pos] = o;
    }
}

// ---------------------------------------------------------------------------
// Bilinear resize [B,32,7,7] -> [B,32,56,56] (half-pixel, clipped coords)
// ---------------------------------------------------------------------------
__global__ __launch_bounds__(256) void resize7to56(const float* __restrict__ t3,
                                                   float* __restrict__ cg)
{
    const int tid = blockIdx.x*256 + threadIdx.x;
    const int ox = tid % WW2;
    const int oy = (tid / WW2) % HH;
    const int ch = (tid / NN) % 32;
    const int b  = tid / (NN*32);
    float sy = ((float)oy + 0.5f)*0.125f - 0.5f; sy = fminf(fmaxf(sy, 0.f), 6.f);
    float sx = ((float)ox + 0.5f)*0.125f - 0.5f; sx = fminf(fmaxf(sx, 0.f), 6.f);
    const int y0 = (int)floorf(sy), x0 = (int)floorf(sx);
    const int y1 = min(y0+1, 6),    x1 = min(x0+1, 6);
    const float wy = sy - (float)y0, wx = sx - (float)x0;
    const float* tb = t3 + ((size_t)b*32 + ch)*49;
    const float r0 = tb[y0*7+x0]*(1.f-wx) + tb[y0*7+x1]*wx;
    const float r1 = tb[y1*7+x0]*(1.f-wx) + tb[y1*7+x1]*wx;
    cg[((size_t)b*32 + ch)*NN + oy*WW2 + ox] = r0*(1.f-wy) + r1*wy;
}

// ---------------------------------------------------------------------------
// Depthwise 3x3, SAME zero pad
// ---------------------------------------------------------------------------
__global__ __launch_bounds__(256) void dwconv3x3(const float* __restrict__ x,
                                                 const float* __restrict__ w,
                                                 float* __restrict__ y)
{
    const int tid = blockIdx.x*256 + threadIdx.x;
    const int px = tid % WW2;
    const int py = (tid / WW2) % HH;
    const int c  = (tid / NN) % CC;
    const int b  = tid / (NN*CC);
    const float* xb = x + ((size_t)b*CC + c)*NN;
    const float* wc = w + c*9;
    float s = 0.f;
#pragma unroll
    for (int dy = -1; dy <= 1; ++dy) {
        const int yy = py + dy;
        if (yy < 0 || yy >= HH) continue;
#pragma unroll
        for (int dx = -1; dx <= 1; ++dx) {
            const int xx = px + dx;
            if (xx < 0 || xx >= WW2) continue;
            s += xb[yy*WW2 + xx] * wc[(dy+1)*3 + (dx+1)];
        }
    }
    y[((size_t)b*CC + c)*NN + py*WW2 + px] = s;
}

// ---------------------------------------------------------------------------
// lo branch: channel-LN + GELU + 32x256 pointwise conv (+bias), per pixel
// ---------------------------------------------------------------------------
__global__ __launch_bounds__(256) void lo_branch(const float* __restrict__ y,
                                                 const float* __restrict__ g,
                                                 const float* __restrict__ be,
                                                 const float* __restrict__ pw,
                                                 const float* __restrict__ pwb,
                                                 float* __restrict__ lo)
{
    const int tid = blockIdx.x*256 + threadIdx.x;
    const int n = tid % NN;
    const int b = tid / NN;
    const float* yb = y + (size_t)b*CC*NN + n;
    float s = 0.f, ss = 0.f;
    for (int c = 0; c < CC; ++c) { const float v = yb[(size_t)c*NN]; s += v; ss += v*v; }
    const float mu  = s * (1.0f/CC);
    float var = ss * (1.0f/CC) - mu*mu;
    var = fmaxf(var, 0.f);
    const float rstd = rsqrtf(var + 1e-6f);
    float acc[32];
#pragma unroll
    for (int o = 0; o < 32; ++o) acc[o] = pwb[o];
    for (int c = 0; c < CC; ++c) {
        const float a = gelu_f((yb[(size_t)c*NN] - mu)*rstd*g[c] + be[c]);
#pragma unroll
        for (int o = 0; o < 32; ++o) acc[o] += pw[o*CC + c] * a;
    }
#pragma unroll
    for (int o = 0; o < 32; ++o) lo[((size_t)b*32 + o)*NN + n] = acc[o];
}

// ---------------------------------------------------------------------------
// 64x64 offset conv -> absolute sample coords. The reference's /W,*2-1,(+1)*0.5*W-0.5
// chain cancels to  gx = x + off_x,  gy = y + off_y.
// ---------------------------------------------------------------------------
__global__ __launch_bounds__(256) void offsets_kernel(const float* __restrict__ cg,
                                                      const float* __restrict__ lo,
                                                      const float* __restrict__ ow,
                                                      const float* __restrict__ ob,
                                                      float* __restrict__ gxp,
                                                      float* __restrict__ gyp)
{
    const int tid = blockIdx.x*256 + threadIdx.x;
    const int n = tid % NN;
    const int b = tid / NN;
    float fused[64];
#pragma unroll
    for (int i = 0; i < 32; ++i) fused[i]    = cg[((size_t)b*32 + i)*NN + n];
#pragma unroll
    for (int i = 0; i < 32; ++i) fused[32+i] = lo[((size_t)b*32 + i)*NN + n];
    const float px = (float)(n % WW2);
    const float py = (float)(n / WW2);
    for (int j = 0; j < 64; ++j) {
        float a = ob[j];
#pragma unroll
        for (int i = 0; i < 64; ++i) a += ow[j*64 + i] * fused[i];
        const int h = j >> 3;
        const int p = (j >> 1) & 3;
        const size_t oi = (((size_t)b*NHD + h)*PP + p)*NN + n;
        if ((j & 1) == 0) gxp[oi] = px + a;
        else              gyp[oi] = py + a;
    }
}

// ---------------------------------------------------------------------------
// Deformable attention core: bilinear sample K/V (f16) at 4 pts, q.k scores,
// analytic relative-position bias, softmax over P=4, attn.V -> out (f32)
// ---------------------------------------------------------------------------
__global__ __launch_bounds__(256) void attn_kernel(const _Float16* __restrict__ q,
                                                   const _Float16* __restrict__ k,
                                                   const _Float16* __restrict__ v,
                                                   const float* __restrict__ gxp,
                                                   const float* __restrict__ gyp,
                                                   const float* __restrict__ biases,
                                                   float* __restrict__ out)
{
    const int tid = blockIdx.x*256 + threadIdx.x;
    const int n = tid % NN;
    const int h = (tid / NN) % NHD;
    const int b = tid / (NN*NHD);
    const int qx = n % WW2, qy = n / WW2;

    int   idxs[PP][4];
    float wts[PP][4];
    float sc[PP];
#pragma unroll
    for (int p = 0; p < PP; ++p) {
        const size_t gi = (((size_t)b*NHD + h)*PP + p)*NN + n;
        const float gx = gxp[gi], gy = gyp[gi];
        const float fx0 = floorf(gx), fy0 = floorf(gy);
        const int x0 = (int)fx0, y0 = (int)fy0;
        const float wx = gx - fx0, wy = gy - fy0;
        const float cw[4] = {(1.f-wx)*(1.f-wy), wx*(1.f-wy), (1.f-wx)*wy, wx*wy};
        const int cx[4] = {x0, x0+1, x0,   x0+1};
        const int cy[4] = {y0, y0,   y0+1, y0+1};
#pragma unroll
        for (int c = 0; c < 4; ++c) {
            const bool valid = (cx[c] >= 0) && (cx[c] < WW2) && (cy[c] >= 0) && (cy[c] < HH);
            const int xi = min(max(cx[c], 0), WW2-1);
            const int yi = min(max(cy[c], 0), HH-1);
            idxs[p][c] = yi*WW2 + xi;
            wts[p][c]  = valid ? cw[c] : 0.f;
        }
        const int bx = min(max((int)rintf(gx), 0), WW2-1);
        const int by = min(max((int)rintf(gy), 0), HH-1);
        const int rel = abs(qy - by)*WW2 + abs(qx - bx);
        sc[p] = biases[h*NN + rel];
    }

    const size_t base = ((size_t)b*CC + h*HD)*NN;
    float dot[PP] = {0.f, 0.f, 0.f, 0.f};
    for (int d = 0; d < HD; ++d) {
        const size_t db = base + (size_t)d*NN;
        const float qd = (float)q[db + n];
#pragma unroll
        for (int p = 0; p < PP; ++p) {
            const float kv = wts[p][0]*(float)k[db + idxs[p][0]]
                           + wts[p][1]*(float)k[db + idxs[p][1]]
                           + wts[p][2]*(float)k[db + idxs[p][2]]
                           + wts[p][3]*(float)k[db + idxs[p][3]];
            dot[p] += qd * kv;
        }
    }
#pragma unroll
    for (int p = 0; p < PP; ++p) sc[p] += dot[p] * SCALE_F;

    float m = sc[0];
#pragma unroll
    for (int p = 1; p < PP; ++p) m = fmaxf(m, sc[p]);
    float a[PP], esum = 0.f;
#pragma unroll
    for (int p = 0; p < PP; ++p) { a[p] = __expf(sc[p] - m); esum += a[p]; }
    const float inv = 1.0f / esum;
#pragma unroll
    for (int p = 0; p < PP; ++p) a[p] *= inv;

    for (int d = 0; d < HD; ++d) {
        const size_t db = base + (size_t)d*NN;
        float od = 0.f;
#pragma unroll
        for (int p = 0; p < PP; ++p) {
            const float vv = wts[p][0]*(float)v[db + idxs[p][0]]
                           + wts[p][1]*(float)v[db + idxs[p][1]]
                           + wts[p][2]*(float)v[db + idxs[p][2]]
                           + wts[p][3]*(float)v[db + idxs[p][3]];
            od += a[p] * vv;
        }
        out[db + n] = od;
    }
}

// ---------------------------------------------------------------------------
extern "C" void kernel_launch(void* const* d_in, const int* in_sizes, int n_in,
                              void* d_out, int out_size, void* d_ws, size_t ws_size,
                              hipStream_t stream)
{
    (void)in_sizes; (void)n_in; (void)out_size; (void)ws_size;
    const float* local_feat    = (const float*)d_in[0];
    const float* context_prior = (const float*)d_in[1];
    const float* deformable_x  = (const float*)d_in[2];
    const float* q_w      = (const float*)d_in[3];
    const float* k_w      = (const float*)d_in[4];
    const float* v_w      = (const float*)d_in[5];
    const float* cg_pre_w = (const float*)d_in[6];
    const float* cg_ln_g  = (const float*)d_in[7];
    const float* cg_ln_b  = (const float*)d_in[8];
    const float* cg_post_w= (const float*)d_in[9];
    const float* lo_dw_w  = (const float*)d_in[10];
    const float* lo_ln_g  = (const float*)d_in[11];
    const float* lo_ln_b  = (const float*)d_in[12];
    const float* lo_pw_w  = (const float*)d_in[13];
    const float* lo_pw_b  = (const float*)d_in[14];
    const float* off_w    = (const float*)d_in[15];
    const float* off_b    = (const float*)d_in[16];
    const float* attn_biases = (const float*)d_in[17];
    const float* proj_w   = (const float*)d_in[18];
    const float* bn_g     = (const float*)d_in[19];
    const float* bn_b     = (const float*)d_in[20];
    const float* bn_mean  = (const float*)d_in[21];
    const float* bn_var   = (const float*)d_in[22];
    // d_in[23] = bias_idxs: recomputed analytically on-device, not read.

    char* wsp = (char*)d_ws;
    size_t off = 0;
    float* bufA   = (float*)(wsp + off);   off += (size_t)BB*CC*NN*4;       // 51.4 MB (reused 3x)
    float* pooled = (float*)(wsp + off);   off += (size_t)BB*CC*49*4;
    float* t3     = (float*)(wsp + off);   off += (size_t)BB*32*49*4;
    float* cg     = (float*)(wsp + off);   off += (size_t)BB*32*NN*4;
    float* lo     = (float*)(wsp + off);   off += (size_t)BB*32*NN*4;
    float* gxp    = (float*)(wsp + off);   off += (size_t)BB*NHD*PP*NN*4;
    float* gyp    = (float*)(wsp + off);   off += (size_t)BB*NHD*PP*NN*4;
    _Float16* qf  = (_Float16*)(wsp + off); off += (size_t)BB*CC*NN*2;
    _Float16* kf  = (_Float16*)(wsp + off); off += (size_t)BB*CC*NN*2;
    _Float16* vf  = (_Float16*)(wsp + off); off += (size_t)BB*CC*NN*2;

    const dim3 gg(4, 49, 16);   // M-tiles, N-tiles, batch

    // cg branch
    gemm256<1, float><<<gg, 256, 0, stream>>>(context_prior, cg_pre_w, bufA,
                                              nullptr, nullptr, nullptr, nullptr);
    pool8x8<<<784, 256, 0, stream>>>(bufA, pooled);
    cg_ln_conv<<<784, 256, 0, stream>>>(pooled, cg_ln_g, cg_ln_b, cg_post_w, t3);
    resize7to56<<<6272, 256, 0, stream>>>(t3, cg);

    // lo branch (bufA reused for dwconv output)
    dwconv3x3<<<50176, 256, 0, stream>>>(local_feat, lo_dw_w, bufA);
    lo_branch<<<196, 256, 0, stream>>>(bufA, lo_ln_g, lo_ln_b, lo_pw_w, lo_pw_b, lo);

    // offsets -> absolute sample coordinates
    offsets_kernel<<<196, 256, 0, stream>>>(cg, lo, off_w, off_b, gxp, gyp);

    // q, k, v projections (f16 outputs)
    gemm256<0, _Float16><<<gg, 256, 0, stream>>>(local_feat,    q_w, qf,
                                                 nullptr, nullptr, nullptr, nullptr);
    gemm256<0, _Float16><<<gg, 256, 0, stream>>>(context_prior, k_w, kf,
                                                 nullptr, nullptr, nullptr, nullptr);
    gemm256<0, _Float16><<<gg, 256, 0, stream>>>(deformable_x,  v_w, vf,
                                                 nullptr, nullptr, nullptr, nullptr);

    // attention (bufA reused for attention output)
    attn_kernel<<<1568, 256, 0, stream>>>(qf, kf, vf, gxp, gyp, attn_biases, bufA);

    // output projection + batchnorm -> d_out
    gemm256<2, float><<<gg, 256, 0, stream>>>(bufA, proj_w, (float*)d_out,
                                              bn_mean, bn_g, bn_var, bn_b);
}
